// Discriminator_56985626084051
// MI455X (gfx1250) — matrix operand order repaired and checked by reading, not compiled, not run on hardware
//
#include <hip/hip_runtime.h>
#include <hip/hip_bf16.h>

// ---------------------------------------------------------------------------
// CDNA5 WMMA types/helpers (gfx1250, wave32)
// ---------------------------------------------------------------------------
typedef __bf16 v16bf __attribute__((ext_vector_type(16)));
typedef __bf16 v8bf  __attribute__((ext_vector_type(8)));
typedef float  v8f   __attribute__((ext_vector_type(8)));

__device__ __forceinline__ v8f wmma_bf16(v16bf a, v16bf b, v8f c) {
  return __builtin_amdgcn_wmma_f32_16x16x32_bf16(false, a, false, b, (short)0, c,
                                                 false, false);
}

// A-fragment (16-bit 16x32): 16 contiguous K values per lane.
// Two aligned 16B loads of bf16, merged in registers (no conversion!).
__device__ __forceinline__ v16bf load_a_frag(const __bf16* __restrict__ p, int off) {
  v8bf lo = *(const v8bf*)(p + off);
  v8bf hi = *(const v8bf*)(p + off + 8);
  return __builtin_shufflevector(lo, hi, 0, 1, 2, 3, 4, 5, 6, 7,
                                 8, 9, 10, 11, 12, 13, 14, 15);
}
// B-fragment (16-bit 32x16): 16 contiguous K values per lane.
__device__ __forceinline__ v16bf load_b_frag(const __bf16* __restrict__ p, int off) {
  v8bf lo = *(const v8bf*)(p + off);
  v8bf hi = *(const v8bf*)(p + off + 8);
  return __builtin_shufflevector(lo, hi, 0, 1, 2, 3, 4, 5, 6, 7,
                                 8, 9, 10, 11, 12, 13, 14, 15);
}

// fp32 -> bf16 convert with K-row padding (pad region zero-filled).
__global__ void cvt_pad_kernel(const float* __restrict__ src, __bf16* __restrict__ dst,
                               int rows, int Ksrc, int Kdst) {
  int i = blockIdx.x * blockDim.x + threadIdx.x;
  if (i >= rows * Kdst) return;
  int r = i / Kdst, k = i - r * Kdst;
  dst[i] = (k < Ksrc) ? (__bf16)src[(size_t)r * Ksrc + k] : (__bf16)0.f;
}

// ---------------------------------------------------------------------------
// Conv2d (NCHW bf16 in, OIHW bf16 weights, K padded to Kpad) as implicit GEMM.
// M=Cout, N=B*Hout*Wout, K=Cin*KH*KW. 32x32 tile per wave (2x2 blocking).
// Branchless loads; padded A columns are zero so B needs only spatial masking.
// ---------------------------------------------------------------------------
template <int KH, int KW>
__global__ void __launch_bounds__(32) conv2d_wmma(
    const __bf16* __restrict__ x, const __bf16* __restrict__ w,
    const float* __restrict__ bias, float* __restrict__ y32, __bf16* __restrict__ y16,
    int Bn, int Cin, int Hin, int Win, int Cout, int Hout, int Wout,
    int stride, int pad, int apply_lrelu, int Kpad) {
  constexpr int KHW = KH * KW;
  const int lane = threadIdx.x;
  const bool half = (lane < 16);
  const int l15 = lane & 15;
  const int m0 = blockIdx.x * 32;
  const int n0 = blockIdx.y * 32;
  const int Ktot = Cin * KHW;
  const int HW = Hout * Wout;

  int nb[2], bb[2], oy[2], ox[2];
#pragma unroll
  for (int t = 0; t < 2; ++t) {
    nb[t] = n0 + t * 16 + l15;
    bb[t] = nb[t] / HW;
    int r = nb[t] % HW;
    oy[t] = r / Wout;
    ox[t] = r % Wout;
  }
  const __bf16* wrow[2];
#pragma unroll
  for (int t = 0; t < 2; ++t) {
    int a = m0 + t * 16 + l15;
    if (a >= Cout) a = Cout - 1;          // clamp: garbage rows never stored
    wrow[t] = w + (size_t)a * Kpad;
  }
  const int abase = half ? 0 : 16;
  const int bbase = half ? 0 : 16;

  const v8f vz = {0.f, 0.f, 0.f, 0.f, 0.f, 0.f, 0.f, 0.f};
  v8f acc[2][2];
#pragma unroll
  for (int i = 0; i < 2; ++i)
#pragma unroll
    for (int j = 0; j < 2; ++j) acc[i][j] = vz;

  for (int k0 = 0; k0 < Kpad; k0 += 32) {
    v16bf A[2], Bf[2];
#pragma unroll
    for (int t = 0; t < 2; ++t) A[t] = load_a_frag(wrow[t], k0 + abase);
#pragma unroll
    for (int t = 0; t < 2; ++t) {
      v16bf f;
#pragma unroll
      for (int i = 0; i < 16; ++i) {
        int k = k0 + bbase + i;
        int kc = k >= Ktot ? Ktot - 1 : k;  // clamp; padded A cols are zero
        int ci = kc / KHW;                  // constant divisor -> mul/shift
        int kr = kc - ci * KHW;
        int ky = kr / KW, kx = kr - ky * KW;
        int iy = oy[t] * stride - pad + ky;
        int ix = ox[t] * stride - pad + kx;
        int iyc = iy < 0 ? 0 : (iy >= Hin ? Hin - 1 : iy);
        int ixc = ix < 0 ? 0 : (ix >= Win ? Win - 1 : ix);
        __bf16 raw = x[(((size_t)bb[t] * Cin + ci) * Hin + iyc) * Win + ixc];
        bool ok = ((unsigned)iy < (unsigned)Hin) && ((unsigned)ix < (unsigned)Win);
        f[i] = ok ? raw : (__bf16)0.f;
      }
      Bf[t] = f;
    }
#pragma unroll
    for (int mt = 0; mt < 2; ++mt)
#pragma unroll
      for (int nt = 0; nt < 2; ++nt) acc[mt][nt] = wmma_bf16(A[mt], Bf[nt], acc[mt][nt]);
  }

#pragma unroll
  for (int mt = 0; mt < 2; ++mt) {
    int mrow = m0 + mt * 16 + (half ? 0 : 8);
#pragma unroll
    for (int nt = 0; nt < 2; ++nt) {
#pragma unroll
      for (int r = 0; r < 8; ++r) {
        int m = mrow + r;
        if (m < Cout) {
          float v = acc[mt][nt][r];
          if (bias) v += bias[m];
          if (apply_lrelu) v = v >= 0.f ? v : 0.2f * v;
          size_t idx = (((size_t)bb[nt] * Cout + m) * Hout + oy[nt]) * Wout + ox[nt];
          if (y32) y32[idx] = v;
          if (y16) y16[idx] = (__bf16)v;
        }
      }
    }
  }
}

// ---------------------------------------------------------------------------
// BatchNorm stats (fp32 in) / apply+LeakyReLU (bf16 out). HW, C powers of two.
// ---------------------------------------------------------------------------
__global__ void bn_stats_kernel(const float* __restrict__ x, float* __restrict__ mean,
                                float* __restrict__ rstd, int Bn, int C, int hw_shift) {
  int c = blockIdx.x;
  const int HW = 1 << hw_shift;
  __shared__ float rs[256], rq[256];
  float s = 0.f, q = 0.f;
  int n = Bn << hw_shift;
  for (int i = threadIdx.x; i < n; i += 256) {
    int b = i >> hw_shift, r = i & (HW - 1);
    float v = x[(((size_t)b * C + c) << hw_shift) + r];
    s += v; q += v * v;
  }
  rs[threadIdx.x] = s; rq[threadIdx.x] = q; __syncthreads();
  for (int o = 128; o > 0; o >>= 1) {
    if (threadIdx.x < o) { rs[threadIdx.x] += rs[threadIdx.x + o]; rq[threadIdx.x] += rq[threadIdx.x + o]; }
    __syncthreads();
  }
  if (threadIdx.x == 0) {
    float m = rs[0] / (float)n;
    float v = rq[0] / (float)n - m * m;
    mean[c] = m;
    rstd[c] = rsqrtf(v + 1e-5f);
  }
}

__global__ void bn_lrelu_kernel(const float* __restrict__ x, const float* __restrict__ mean,
                                const float* __restrict__ rstd, const float* __restrict__ g,
                                const float* __restrict__ b2, __bf16* __restrict__ y,
                                int hw_shift, int cmask, long n) {
  long i = (long)blockIdx.x * blockDim.x + threadIdx.x;
  if (i >= n) return;
  int c = (int)(i >> hw_shift) & cmask;
  float v = (x[i] - mean[c]) * rstd[c] * g[c] + b2[c];
  y[i] = (__bf16)(v >= 0.f ? v : 0.2f * v);
}

__global__ void zero_kernel(float* p, int n) {
  int i = blockIdx.x * blockDim.x + threadIdx.x;
  if (i < n) p[i] = 0.f;
}

// ---------------------------------------------------------------------------
// GRU input projection, hoisted for ALL timesteps/directions:
// gi[dir][(l*B+b), :] = text[b, dir? L-1-l : l, :] @ Wih^T + bih.
// M=1536, N=1024, K=320 (padded). bf16 operands, 10 clean k-steps, no tail.
// ---------------------------------------------------------------------------
__global__ void __launch_bounds__(32) gru_gi_kernel(
    const __bf16* __restrict__ textb, const __bf16* __restrict__ wihb,
    const float* __restrict__ bih_f, const float* __restrict__ bih_b,
    float* __restrict__ gi_f, float* __restrict__ gi_b) {
  const int dir = blockIdx.z;
  const __bf16* wih = wihb + (size_t)dir * 1536 * 320;
  const float* bih = dir ? bih_b : bih_f;
  float* gi = dir ? gi_b : gi_f;
  const int lane = threadIdx.x;
  const bool half = (lane < 16);
  const int l15 = lane & 15;
  const int m0 = blockIdx.x * 32;
  const int n0 = blockIdx.y * 32;
  const int abase = half ? 0 : 16;
  const int bbase = half ? 0 : 16;

  const __bf16* xrow[2];
  const __bf16* wrow[2];
  int nb[2];
#pragma unroll
  for (int t = 0; t < 2; ++t) {
    nb[t] = n0 + t * 16 + l15;
    int lrow = nb[t] >> 5, bcol = nb[t] & 31;
    int lsrc = dir ? (31 - lrow) : lrow;
    xrow[t] = textb + ((size_t)bcol * 32 + lsrc) * 320;
    wrow[t] = wih + (size_t)(m0 + t * 16 + l15) * 320;
  }

  const v8f vz = {0.f, 0.f, 0.f, 0.f, 0.f, 0.f, 0.f, 0.f};
  v8f acc[2][2];
#pragma unroll
  for (int i = 0; i < 2; ++i)
#pragma unroll
    for (int j = 0; j < 2; ++j) acc[i][j] = vz;

  for (int k0 = 0; k0 < 320; k0 += 32) {
    v16bf A[2], Bf[2];
#pragma unroll
    for (int t = 0; t < 2; ++t) A[t] = load_a_frag(wrow[t], k0 + abase);
#pragma unroll
    for (int t = 0; t < 2; ++t) Bf[t] = load_b_frag(xrow[t], k0 + bbase);
#pragma unroll
    for (int mt = 0; mt < 2; ++mt)
#pragma unroll
      for (int nt = 0; nt < 2; ++nt) acc[mt][nt] = wmma_bf16(A[mt], Bf[nt], acc[mt][nt]);
  }

#pragma unroll
  for (int mt = 0; mt < 2; ++mt) {
    int mrow = m0 + mt * 16 + (half ? 0 : 8);
    const float4 bi0 = *(const float4*)(bih + mrow);
    const float4 bi1 = *(const float4*)(bih + mrow + 4);
#pragma unroll
    for (int nt = 0; nt < 2; ++nt) {
      float4 o0, o1;
      o0.x = acc[mt][nt][0] + bi0.x; o0.y = acc[mt][nt][1] + bi0.y;
      o0.z = acc[mt][nt][2] + bi0.z; o0.w = acc[mt][nt][3] + bi0.w;
      o1.x = acc[mt][nt][4] + bi1.x; o1.y = acc[mt][nt][5] + bi1.y;
      o1.z = acc[mt][nt][6] + bi1.z; o1.w = acc[mt][nt][7] + bi1.w;
      float* dst = gi + (size_t)nb[nt] * 1536 + mrow;
      *(float4*)(dst) = o0;
      *(float4*)(dst + 4) = o1;
    }
  }
}

// ---------------------------------------------------------------------------
// Recurrent GEMM: gh[dir][b,:] = hbf[dir][b,:] @ Whh^T + bhh. M=1536,N=32,K=512.
// ---------------------------------------------------------------------------
__global__ void __launch_bounds__(32) gru_gh_kernel(
    const __bf16* __restrict__ hbf, const __bf16* __restrict__ whhb,
    const float* __restrict__ bhh_f, const float* __restrict__ bhh_b,
    float* __restrict__ gh) {
  const int dir = blockIdx.z;
  const __bf16* whh = whhb + (size_t)dir * 1536 * 512;
  const float* bhh = dir ? bhh_b : bhh_f;
  const __bf16* hp = hbf + (size_t)dir * 32 * 512;
  float* ghp = gh + (size_t)dir * 32 * 1536;
  const int lane = threadIdx.x;
  const bool half = (lane < 16);
  const int l15 = lane & 15;
  const int m0 = blockIdx.x * 32;
  const int abase = half ? 0 : 16;
  const int bbase = half ? 0 : 16;

  const __bf16* xrow[2];
  const __bf16* wrow[2];
  int nb[2];
#pragma unroll
  for (int t = 0; t < 2; ++t) {
    nb[t] = t * 16 + l15;
    xrow[t] = hp + (size_t)nb[t] * 512;
    wrow[t] = whh + (size_t)(m0 + t * 16 + l15) * 512;
  }

  const v8f vz = {0.f, 0.f, 0.f, 0.f, 0.f, 0.f, 0.f, 0.f};
  v8f acc[2][2];
#pragma unroll
  for (int i = 0; i < 2; ++i)
#pragma unroll
    for (int j = 0; j < 2; ++j) acc[i][j] = vz;

  for (int k0 = 0; k0 < 512; k0 += 32) {
    v16bf A[2], Bf[2];
#pragma unroll
    for (int t = 0; t < 2; ++t) A[t] = load_a_frag(wrow[t], k0 + abase);
#pragma unroll
    for (int t = 0; t < 2; ++t) Bf[t] = load_b_frag(xrow[t], k0 + bbase);
#pragma unroll
    for (int mt = 0; mt < 2; ++mt)
#pragma unroll
      for (int nt = 0; nt < 2; ++nt) acc[mt][nt] = wmma_bf16(A[mt], Bf[nt], acc[mt][nt]);
  }

#pragma unroll
  for (int mt = 0; mt < 2; ++mt) {
    int mrow = m0 + mt * 16 + (half ? 0 : 8);
    const float4 bi0 = *(const float4*)(bhh + mrow);
    const float4 bi1 = *(const float4*)(bhh + mrow + 4);
#pragma unroll
    for (int nt = 0; nt < 2; ++nt) {
      float4 o0, o1;
      o0.x = acc[mt][nt][0] + bi0.x; o0.y = acc[mt][nt][1] + bi0.y;
      o0.z = acc[mt][nt][2] + bi0.z; o0.w = acc[mt][nt][3] + bi0.w;
      o1.x = acc[mt][nt][4] + bi1.x; o1.y = acc[mt][nt][5] + bi1.y;
      o1.z = acc[mt][nt][6] + bi1.z; o1.w = acc[mt][nt][7] + bi1.w;
      float* dst = ghp + (size_t)nb[nt] * 1536 + mrow;
      *(float4*)(dst) = o0;
      *(float4*)(dst + 4) = o1;
    }
  }
}

// Gate nonlinearities + state update (fp32 state + bf16 mirror) + masked output.
__global__ void gru_gate_kernel(const float* __restrict__ gi_f, const float* __restrict__ gi_b,
                                const float* __restrict__ gh, float* __restrict__ h,
                                __bf16* __restrict__ hbf,
                                float* __restrict__ hf_mat, float* __restrict__ hb_raw,
                                const int* __restrict__ len_text, int l) {
  int idx = blockIdx.x * blockDim.x + threadIdx.x;  // 2*32*512
  if (idx >= 2 * 32 * 512) return;
  int dir = idx >> 14;
  int r = idx & 16383;
  int b = r >> 9, j = r & 511;
  const float* gi = (dir ? gi_b : gi_f) + ((size_t)l * 32 + b) * 1536;
  const float* ghp = gh + ((size_t)dir * 32 + b) * 1536;
  float ir = gi[j], iz = gi[j + 512], inn = gi[j + 1024];
  float hr = ghp[j], hz = ghp[j + 512], hn = ghp[j + 1024];
  float rg = 1.f / (1.f + __expf(-(ir + hr)));
  float zg = 1.f / (1.f + __expf(-(iz + hz)));
  float ng = tanhf(inn + rg * hn);
  size_t hidx = ((size_t)dir * 32 + b) * 512 + j;
  float hprev = h[hidx];
  float hnew = (1.f - zg) * ng + zg * hprev;
  h[hidx] = hnew;
  hbf[hidx] = (__bf16)hnew;
  float m = (l < len_text[b]) ? 1.f : 0.f;
  float out = hnew * m;
  size_t oidx = ((size_t)l * 32 + b) * 512 + j;
  if (dir == 0) hf_mat[oidx] = out; else hb_raw[oidx] = out;
}

// words[b,l,:] = 0.5*(hf_mat[l,b,:] + hb_raw[(l? L-l : 0),b,:])
__global__ void words_kernel(const float* __restrict__ hf, const float* __restrict__ hb,
                             float* __restrict__ words) {
  int idx = blockIdx.x * blockDim.x + threadIdx.x;
  if (idx >= 32 * 32 * 512) return;
  int b = idx >> 14;
  int r = idx & 16383;
  int l = r >> 9, j = r & 511;
  int ls = (l == 0) ? 0 : (32 - l);
  words[idx] = 0.5f * (hf[((size_t)l * 32 + b) * 512 + j] +
                       hb[((size_t)ls * 32 + b) * 512 + j]);
}

__global__ void avg_kernel(const float* __restrict__ words, const int* __restrict__ len_text,
                           float* __restrict__ avg) {
  int b = blockIdx.x, j = threadIdx.x;  // blockDim = 512
  float s = 0.f;
  for (int l = 0; l < 32; ++l) s += words[((size_t)b * 32 + l) * 512 + j];
  avg[b * 512 + j] = s / (float)len_text[b];
}

__global__ void score_kernel(const float* __restrict__ words, const float* __restrict__ avg,
                             float* __restrict__ sc) {
  int bl = blockIdx.x;
  int b = bl >> 5;
  __shared__ float red[128];
  float s = 0.f;
  for (int j = threadIdx.x; j < 512; j += 128)
    s += words[(size_t)bl * 512 + j] * avg[b * 512 + j];
  red[threadIdx.x] = s; __syncthreads();
  for (int o = 64; o > 0; o >>= 1) {
    if (threadIdx.x < o) red[threadIdx.x] += red[threadIdx.x + o];
    __syncthreads();
  }
  if (threadIdx.x == 0) sc[bl] = red[0];
}

__global__ void softmax_l_kernel(const float* __restrict__ sc, float* __restrict__ alphas) {
  int b = blockIdx.x, l = threadIdx.x;  // blockDim = 32
  __shared__ float sm[32];
  float v = sc[b * 32 + l];
  sm[l] = v; __syncthreads();
  float mx = -1e30f;
  for (int i = 0; i < 32; ++i) mx = fmaxf(mx, sm[i]);
  __syncthreads();
  float e = __expf(v - mx);
  sm[l] = e; __syncthreads();
  float s = 0.f;
  for (int i = 0; i < 32; ++i) s += sm[i];
  alphas[b * 32 + l] = e / s;
}

__global__ void betas_kernel(const float* __restrict__ words, const float* __restrict__ bw,
                             const float* __restrict__ bb, float* __restrict__ betas) {
  int bl = blockIdx.x;
  int g = threadIdx.x >> 5, t = threadIdx.x & 31;  // blockDim = 96
  __shared__ float sm[3][32];
  float s = 0.f;
  for (int j = t; j < 512; j += 32) s += words[(size_t)bl * 512 + j] * bw[g * 512 + j];
  sm[g][t] = s; __syncthreads();
  if (threadIdx.x < 3) {
    float tot = 0.f;
    for (int i = 0; i < 32; ++i) tot += sm[threadIdx.x][i];
    sm[threadIdx.x][0] = tot + bb[threadIdx.x];
  }
  __syncthreads();
  if (threadIdx.x == 0) {
    float a = sm[0][0], b2 = sm[1][0], c = sm[2][0];
    float mx = fmaxf(a, fmaxf(b2, c));
    float ea = __expf(a - mx), eb = __expf(b2 - mx), ec = __expf(c - mx);
    float tt = ea + eb + ec;
    betas[bl * 3 + 0] = ea / tt;
    betas[bl * 3 + 1] = eb / tt;
    betas[bl * 3 + 2] = ec / tt;
  }
}

__global__ void chmean_kernel(const __bf16* __restrict__ x, float* __restrict__ out, int HW) {
  int bc = blockIdx.x;
  __shared__ float red[64];
  float s = 0.f;
  for (int i = threadIdx.x; i < HW; i += 64) s += (float)x[(size_t)bc * HW + i];
  red[threadIdx.x] = s; __syncthreads();
  for (int o = 32; o > 0; o >>= 1) {
    if (threadIdx.x < o) red[threadIdx.x] += red[threadIdx.x + o];
    __syncthreads();
  }
  if (threadIdx.x == 0) out[bc] = red[0] / (float)HW;
}

// U[b,h] = sum_c wbw[c,h]*img[b,c] + wbw[C,h];  t0[b] = sum_c wbb[c]*img[b,c] + wbb[C]
__global__ void computeU_kernel(const float* __restrict__ wbw, const float* __restrict__ wbb,
                                const float* __restrict__ img, float* __restrict__ Ubuf,
                                float* __restrict__ t0, int C) {
  int b = blockIdx.x, j = threadIdx.x;  // blockDim = 512
  float s = wbw[(size_t)C * 512 + j];
  for (int c = 0; c < C; ++c) s += wbw[(size_t)c * 512 + j] * img[b * C + c];
  Ubuf[b * 512 + j] = s;
  if (j == 0) {
    float s0 = wbb[C];
    for (int c = 0; c < C; ++c) s0 += wbb[c] * img[b * C + c];
    t0[b] = s0;
  }
}

// pos[b,l] += sigmoid(words[b,l].U[b] + t0[b]) * betas[b,l,j]
// neg[b,l] += sigmoid(words[bn,l].U[b] + t0[b]) * betas[bn,l,j],  bn=(b-1) mod B
__global__ void posneg_kernel(const float* __restrict__ words, const float* __restrict__ Ubuf,
                              const float* __restrict__ t0, const float* __restrict__ betas,
                              float* __restrict__ pos, float* __restrict__ neg, int j) {
  int bl = blockIdx.x;
  int b = bl >> 5, l = bl & 31;
  int bn = (b == 0) ? 31 : (b - 1);
  __shared__ float red[2][128];
  float sp = 0.f, sn = 0.f;
  for (int h = threadIdx.x; h < 512; h += 128) {
    float u = Ubuf[b * 512 + h];
    sp += words[((size_t)b * 32 + l) * 512 + h] * u;
    sn += words[((size_t)bn * 32 + l) * 512 + h] * u;
  }
  red[0][threadIdx.x] = sp; red[1][threadIdx.x] = sn; __syncthreads();
  for (int o = 64; o > 0; o >>= 1) {
    if (threadIdx.x < o) {
      red[0][threadIdx.x] += red[0][threadIdx.x + o];
      red[1][threadIdx.x] += red[1][threadIdx.x + o];
    }
    __syncthreads();
  }
  if (threadIdx.x == 0) {
    float f = 1.f / (1.f + __expf(-(red[0][0] + t0[b])));
    float fn = 1.f / (1.f + __expf(-(red[1][0] + t0[b])));
    pos[bl] += f * betas[bl * 3 + j];
    neg[bl] += fn * betas[(bn * 32 + l) * 3 + j];
  }
}

__global__ void final_kernel(const float* __restrict__ uncond, const float* __restrict__ alphas,
                             const float* __restrict__ pos, const float* __restrict__ neg,
                             float* __restrict__ out) {
  int b = threadIdx.x;  // blockDim = 32
  int bn = (b == 0) ? 31 : (b - 1);
  float cp = 0.f, cn = 0.f;
  for (int l = 0; l < 32; ++l) {
    cp += alphas[b * 32 + l] * fminf(logf(pos[b * 32 + l] + 1e-8f), 0.f);
    cn += alphas[bn * 32 + l] * fminf(logf(neg[b * 32 + l] + 1e-8f), 0.f);
  }
  out[b] = uncond[b];
  out[32 + b] = cp;
  out[64 + b] = cn;
}

// ---------------------------------------------------------------------------
// Host orchestration
// ---------------------------------------------------------------------------
extern "C" void kernel_launch(void* const* d_in, const int* in_sizes, int n_in,
                              void* d_out, int out_size, void* d_ws, size_t ws_size,
                              hipStream_t stream) {
  const float* image = (const float*)d_in[0];
  const float* text  = (const float*)d_in[1];
  const float* c1w = (const float*)d_in[2];  const float* c1b = (const float*)d_in[3];
  const float* c2w = (const float*)d_in[4];
  const float* bn2g = (const float*)d_in[5]; const float* bn2b = (const float*)d_in[6];
  const float* c3w = (const float*)d_in[7];
  const float* bn3g = (const float*)d_in[8]; const float* bn3b = (const float*)d_in[9];
  const float* c4w = (const float*)d_in[10];
  const float* bn4g = (const float*)d_in[11]; const float* bn4b = (const float*)d_in[12];
  const float* c5w = (const float*)d_in[13];
  const float* bn5g = (const float*)d_in[14]; const float* bn5b = (const float*)d_in[15];
  const float* g3w = (const float*)d_in[16];
  const float* gbn3g = (const float*)d_in[17]; const float* gbn3b = (const float*)d_in[18];
  const float* g4w = (const float*)d_in[19];
  const float* gbn4g = (const float*)d_in[20]; const float* gbn4b = (const float*)d_in[21];
  const float* g5w = (const float*)d_in[22];
  const float* gbn5g = (const float*)d_in[23]; const float* gbn5b = (const float*)d_in[24];
  const float* unw = (const float*)d_in[25]; const float* unb = (const float*)d_in[26];
  const float* betw = (const float*)d_in[27]; const float* betb = (const float*)d_in[28];
  const float* f_wih = (const float*)d_in[29]; const float* f_whh = (const float*)d_in[30];
  const float* f_bih = (const float*)d_in[31]; const float* f_bhh = (const float*)d_in[32];
  const float* b_wih = (const float*)d_in[33]; const float* b_whh = (const float*)d_in[34];
  const float* b_bih = (const float*)d_in[35]; const float* b_bhh = (const float*)d_in[36];
  const float* wb1w = (const float*)d_in[37]; const float* wb1b = (const float*)d_in[38];
  const float* wb2w = (const float*)d_in[39]; const float* wb2b = (const float*)d_in[40];
  const float* wb3w = (const float*)d_in[41]; const float* wb3b = (const float*)d_in[42];
  const int* len_text = (const int*)d_in[43];

  // ---- workspace carve-up (byte allocator, 32B-aligned chunks) ----
  char* wsb = (char*)d_ws;
  size_t off = 0;
  auto allocf = [&](size_t n) { float* p = (float*)(wsb + off);
                                off = (off + n * 4 + 31) & ~(size_t)31; return p; };
  auto alloch = [&](size_t n) { __bf16* p = (__bf16*)(wsb + off);
                                off = (off + n * 2 + 31) & ~(size_t)31; return p; };
  // fp32 scratch
  float* tmp  = allocf(4194304);   // pre-BN conv out (max = c2)
  float* gi_f = allocf(1572864);
  float* gi_b = allocf(1572864);
  float* hf_m = allocf(524288);
  float* hb_r = allocf(524288);
  float* wrd  = allocf(524288);
  float* hbuf = allocf(32768);     // (2,32,512) fp32 recurrent state
  float* ghb  = allocf(98304);     // (2,32,1536)
  float* bn_mean = allocf(512);
  float* bn_rstd = allocf(512);
  float* unc  = allocf(32);
  float* avg  = allocf(16384);
  float* sc   = allocf(1024);
  float* alph = allocf(1024);
  float* bet  = allocf(3072);
  float* img3 = allocf(8192);
  float* img4 = allocf(16384);
  float* img5 = allocf(16384);
  float* Ub   = allocf(16384);
  float* t0b  = allocf(32);
  float* pos  = allocf(1024);
  float* neg  = allocf(1024);
  // bf16 activations
  __bf16* imageb = alloch(1572864);   // 32*3*128*128
  __bf16* x1b = alloch(8388608);
  __bf16* x2b = alloch(4194304);
  __bf16* x3b = alloch(2097152);
  __bf16* x4b = alloch(524288);
  __bf16* x5b = alloch(131072);
  __bf16* gp3b = alloch(2097152);
  __bf16* gp4b = alloch(1048576);
  __bf16* gp5b = alloch(131072);
  __bf16* hbfb = alloch(32768);
  // bf16 weights (K-padded)
  __bf16* c1wb = alloch(64 * 64);          // 48 -> 64
  __bf16* c2wb = alloch(128 * 1024);
  __bf16* c3wb = alloch(256 * 2048);
  __bf16* c4wb = alloch(512 * 4096);
  __bf16* c5wb = alloch(512 * 8192);
  __bf16* g3wb = alloch(256 * 2304);
  __bf16* g4wb = alloch(512 * 4608);
  __bf16* g5wb = alloch(512 * 4608);
  __bf16* unwb = alloch(8192);
  __bf16* textb = alloch(1024 * 320);      // 300 -> 320
  __bf16* wihb = alloch(2 * 1536 * 320);   // fwd,bwd
  __bf16* whhb = alloch(2 * 1536 * 512);

  // ---- one-time bf16 conversion / padding ----
  auto cvt = [&](const float* s, __bf16* d, int rows, int Ks, int Kd) {
    int n = rows * Kd;
    cvt_pad_kernel<<<(n + 255) / 256, 256, 0, stream>>>(s, d, rows, Ks, Kd);
  };
  cvt(image, imageb, 1, 1572864, 1572864);
  cvt(c1w, c1wb, 64, 48, 64);
  cvt(c2w, c2wb, 128, 1024, 1024);
  cvt(c3w, c3wb, 256, 2048, 2048);
  cvt(c4w, c4wb, 512, 4096, 4096);
  cvt(c5w, c5wb, 512, 8192, 8192);
  cvt(g3w, g3wb, 256, 2304, 2304);
  cvt(g4w, g4wb, 512, 4608, 4608);
  cvt(g5w, g5wb, 512, 4608, 4608);
  cvt(unw, unwb, 1, 8192, 8192);
  cvt(text, textb, 1024, 300, 320);
  cvt(f_wih, wihb, 1536, 300, 320);
  cvt(b_wih, wihb + (size_t)1536 * 320, 1536, 300, 320);
  cvt(f_whh, whhb, 1536, 512, 512);
  cvt(b_whh, whhb + (size_t)1536 * 512, 1536, 512, 512);

  // ---- CNN encoder ----
  conv2d_wmma<4, 4><<<dim3(2, 4096), 32, 0, stream>>>(
      imageb, c1wb, c1b, nullptr, x1b, 32, 3, 128, 128, 64, 64, 64, 2, 1, 1, 64);
  conv2d_wmma<4, 4><<<dim3(4, 1024), 32, 0, stream>>>(
      x1b, c2wb, nullptr, tmp, nullptr, 32, 64, 64, 64, 128, 32, 32, 2, 1, 0, 1024);
  bn_stats_kernel<<<128, 256, 0, stream>>>(tmp, bn_mean, bn_rstd, 32, 128, 10);
  bn_lrelu_kernel<<<16384, 256, 0, stream>>>(tmp, bn_mean, bn_rstd, bn2g, bn2b, x2b, 10, 127, 4194304L);
  conv2d_wmma<4, 4><<<dim3(8, 256), 32, 0, stream>>>(
      x2b, c3wb, nullptr, tmp, nullptr, 32, 128, 32, 32, 256, 16, 16, 2, 1, 0, 2048);
  bn_stats_kernel<<<256, 256, 0, stream>>>(tmp, bn_mean, bn_rstd, 32, 256, 8);
  bn_lrelu_kernel<<<8192, 256, 0, stream>>>(tmp, bn_mean, bn_rstd, bn3g, bn3b, x3b, 8, 255, 2097152L);
  conv2d_wmma<4, 4><<<dim3(16, 64), 32, 0, stream>>>(
      x3b, c4wb, nullptr, tmp, nullptr, 32, 256, 16, 16, 512, 8, 8, 2, 1, 0, 4096);
  bn_stats_kernel<<<512, 256, 0, stream>>>(tmp, bn_mean, bn_rstd, 32, 512, 6);
  bn_lrelu_kernel<<<2048, 256, 0, stream>>>(tmp, bn_mean, bn_rstd, bn4g, bn4b, x4b, 6, 511, 524288L);
  conv2d_wmma<4, 4><<<dim3(16, 16), 32, 0, stream>>>(
      x4b, c5wb, nullptr, tmp, nullptr, 32, 512, 8, 8, 512, 4, 4, 2, 1, 0, 8192);
  bn_stats_kernel<<<512, 256, 0, stream>>>(tmp, bn_mean, bn_rstd, 32, 512, 4);
  bn_lrelu_kernel<<<512, 256, 0, stream>>>(tmp, bn_mean, bn_rstd, bn5g, bn5b, x5b, 4, 511, 131072L);
  conv2d_wmma<3, 3><<<dim3(8, 256), 32, 0, stream>>>(
      x3b, g3wb, nullptr, tmp, nullptr, 32, 256, 16, 16, 256, 16, 16, 1, 1, 0, 2304);
  bn_stats_kernel<<<256, 256, 0, stream>>>(tmp, bn_mean, bn_rstd, 32, 256, 8);
  bn_lrelu_kernel<<<8192, 256, 0, stream>>>(tmp, bn_mean, bn_rstd, gbn3g, gbn3b, gp3b, 8, 255, 2097152L);
  conv2d_wmma<3, 3><<<dim3(16, 64), 32, 0, stream>>>(
      x4b, g4wb, nullptr, tmp, nullptr, 32, 512, 8, 8, 512, 8, 8, 1, 1, 0, 4608);
  bn_stats_kernel<<<512, 256, 0, stream>>>(tmp, bn_mean, bn_rstd, 32, 512, 6);
  bn_lrelu_kernel<<<2048, 256, 0, stream>>>(tmp, bn_mean, bn_rstd, gbn4g, gbn4b, gp4b, 6, 511, 524288L);
  conv2d_wmma<3, 3><<<dim3(16, 16), 32, 0, stream>>>(
      x5b, g5wb, nullptr, tmp, nullptr, 32, 512, 4, 4, 512, 4, 4, 1, 1, 0, 4608);
  bn_stats_kernel<<<512, 256, 0, stream>>>(tmp, bn_mean, bn_rstd, 32, 512, 4);
  bn_lrelu_kernel<<<512, 256, 0, stream>>>(tmp, bn_mean, bn_rstd, gbn5g, gbn5b, gp5b, 4, 511, 131072L);
  conv2d_wmma<4, 4><<<dim3(1, 1), 32, 0, stream>>>(
      gp5b, unwb, unb, unc, nullptr, 32, 512, 4, 4, 1, 1, 1, 1, 0, 0, 8192);

  // ---- GRU ----
  zero_kernel<<<128, 256, 0, stream>>>(hbuf, 2 * 32 * 512);
  zero_kernel<<<64, 256, 0, stream>>>((float*)hbfb, 16384);  // bf16 zeros
  zero_kernel<<<8, 256, 0, stream>>>(pos, 1024);
  zero_kernel<<<8, 256, 0, stream>>>(neg, 1024);
  gru_gi_kernel<<<dim3(48, 32, 2), 32, 0, stream>>>(textb, wihb, f_bih, b_bih, gi_f, gi_b);
  for (int l = 0; l < 32; ++l) {
    gru_gh_kernel<<<dim3(48, 1, 2), 32, 0, stream>>>(hbfb, whhb, f_bhh, b_bhh, ghb);
    gru_gate_kernel<<<128, 256, 0, stream>>>(gi_f, gi_b, ghb, hbuf, hbfb, hf_m, hb_r, len_text, l);
  }
  words_kernel<<<2048, 256, 0, stream>>>(hf_m, hb_r, wrd);

  // ---- attention / scoring ----
  avg_kernel<<<32, 512, 0, stream>>>(wrd, len_text, avg);
  score_kernel<<<1024, 128, 0, stream>>>(wrd, avg, sc);
  softmax_l_kernel<<<32, 32, 0, stream>>>(sc, alph);
  betas_kernel<<<1024, 96, 0, stream>>>(wrd, betw, betb, bet);
  chmean_kernel<<<32 * 256, 64, 0, stream>>>(gp3b, img3, 256);
  chmean_kernel<<<32 * 512, 64, 0, stream>>>(gp4b, img4, 64);
  chmean_kernel<<<32 * 512, 64, 0, stream>>>(gp5b, img5, 16);

  const float* wbw[3] = {wb1w, wb2w, wb3w};
  const float* wbb[3] = {wb1b, wb2b, wb3b};
  const float* imgs[3] = {img3, img4, img5};
  const int Cs[3] = {256, 512, 512};
  for (int j = 0; j < 3; ++j) {
    computeU_kernel<<<32, 512, 0, stream>>>(wbw[j], wbb[j], imgs[j], Ub, t0b, Cs[j]);
    posneg_kernel<<<1024, 128, 0, stream>>>(wrd, Ub, t0b, bet, pos, neg, j);
  }
  final_kernel<<<1, 32, 0, stream>>>(unc, alph, pos, neg, (float*)d_out);
}
